// MultiheadEnergyAttention_65506841198694
// MI455X (gfx1250) — matrix operand order repaired and checked by reading, not compiled, not run on hardware
//
#include <hip/hip_runtime.h>
#include <hip/hip_bf16.h>

// ---------------------------------------------------------------------------
// MultiheadEnergyAttention energy for MI455X (gfx1250), wave32 + WMMA bf16.
//   x:[4,2048,768] f32, Wq/Wk:[12,64,768] f32, beta:[12] f32  ->  scalar f32
// Pipeline:
//   f32->bf16 convert
//   Q/K projection GEMM, 32x32 register-blocked v_wmma_f32_16x16x32_bf16
//     (beta folded into Q at store time)
//   fused score-GEMM + online LSE: 32 queries/wave, 1 b128 load per WMMA,
//     diagonal self-exclusion hoisted out of the key loop
//   deterministic tree reduce.
// ---------------------------------------------------------------------------

typedef __bf16 bf16_t;
typedef __attribute__((ext_vector_type(16))) __bf16 v16bf;
typedef __attribute__((ext_vector_type(8)))  __bf16 v8bf;
typedef __attribute__((ext_vector_type(4)))  __bf16 v4bf;
typedef __attribute__((ext_vector_type(8)))  float  v8f;

union Frag16 { v16bf v; v8bf h[2]; };

// Problem constants
constexpr int Bb = 4;
constexpr int Ss = 2048;
constexpr int Ee = 768;
constexpr int Hh = 12;
constexpr int Dd = 64;
constexpr int Mrows = Bb * Ss;                 // 8192  (GEMM M)
constexpr int Ncols = Hh * Dd;                 // 768   (GEMM N)
constexpr int NWAVES_ATTN = (Bb * Hh * Ss) / 32;   // 3072 (32 queries/wave)

// Workspace layout (bytes)
constexpr size_t OFF_XB   = 0;
constexpr size_t SZ_XB    = (size_t)Mrows * Ee * 2;                 // 12.6 MB
constexpr size_t OFF_WQB  = OFF_XB  + SZ_XB;
constexpr size_t SZ_WB    = (size_t)Ncols * Ee * 2;                 // 1.2 MB
constexpr size_t OFF_WKB  = OFF_WQB + SZ_WB;
constexpr size_t OFF_QB   = OFF_WKB + SZ_WB;
constexpr size_t SZ_QKB   = (size_t)Bb * Hh * Ss * Dd * 2;          // 12.6 MB
constexpr size_t OFF_KB   = OFF_QB  + SZ_QKB;
constexpr size_t OFF_PART = OFF_KB  + SZ_QKB;                       // 3072 f32

#define WMMA_BF16(a, b, c) \
  __builtin_amdgcn_wmma_f32_16x16x32_bf16(false, (a).v, false, (b).v, \
                                          (short)0, (c), false, false)

// ---------------------------------------------------------------------------
// f32 -> bf16, 4 elements per thread (element counts are all multiples of 4)
// ---------------------------------------------------------------------------
__global__ void cvt_f32_to_bf16_x4(const float* __restrict__ in,
                                   bf16_t* __restrict__ out, int n4) {
  int i = blockIdx.x * blockDim.x + threadIdx.x;
  if (i >= n4) return;
  float4 f = reinterpret_cast<const float4*>(in)[i];
  v4bf o;
  o.x = (bf16_t)f.x; o.y = (bf16_t)f.y; o.z = (bf16_t)f.z; o.w = (bf16_t)f.w;
  reinterpret_cast<v4bf*>(out)[i] = o;
}

// ---------------------------------------------------------------------------
// Store one 16x16 f32 accumulator tile (optionally scaled per-head) as bf16
// into [B,H,S,D].  C/D layout: lane holds column n = lane%16; rows
// m = r + 8*(lane>>4).
// ---------------------------------------------------------------------------
__device__ __forceinline__ void store_tile(const v8f& c, int mbase, int nbase,
                                           int l15, int kh,
                                           const float* __restrict__ scale,
                                           bf16_t* __restrict__ outb) {
  const int n  = nbase + l15;
  const int h  = n >> 6;
  const int d  = n & 63;
  const float sc = scale ? scale[h] : 1.0f;
  const int m0 = mbase + 8 * kh;
#pragma unroll
  for (int r = 0; r < 8; ++r) {
    const int m  = m0 + r;
    const int bi = m >> 11;      // /2048
    const int s  = m & 2047;
    outb[(((size_t)bi * Hh + h) * Ss + s) * Dd + d] = (bf16_t)(c[r] * sc);
  }
}

// ---------------------------------------------------------------------------
// Projection GEMM:  out[b,h,s,d] = scale_h * sum_e x[b,s,e] * W[h,d,e].
// 32x32 output tile per wave: 2 A-frags x 2 B-frags -> 4 WMMAs per K-step,
// 8 b128 loads per 4 WMMAs (2 loads/WMMA).
// ---------------------------------------------------------------------------
__global__ void __launch_bounds__(128)
proj_gemm(const bf16_t* __restrict__ xb,    // [8192,768]
          const bf16_t* __restrict__ wb,    // [768,768]  (row n = h*64+d)
          const float* __restrict__ scale,  // [H] or nullptr
          bf16_t* __restrict__ outb)        // [B,H,S,D]
{
  const int lane  = threadIdx.x & 31;
  const int wid   = threadIdx.x >> 5;
  const int mtile = blockIdx.x * 4 + wid;      // 0..255 (32-row tiles)
  const int ntile = blockIdx.y;                // 0..23  (32-col tiles)
  const int l15   = lane & 15;
  const int kh    = lane >> 4;                 // half-wave selector

  const bf16_t* arow0 = xb + (size_t)(mtile * 32 + l15) * Ee;
  const bf16_t* arow1 = arow0 + (size_t)16 * Ee;
  const bf16_t* brow0 = wb + (size_t)(ntile * 32 + l15) * Ee;
  const bf16_t* brow1 = brow0 + (size_t)16 * Ee;

  v8f c00 = {}, c01 = {}, c10 = {}, c11 = {};
#pragma unroll 2
  for (int kk = 0; kk < Ee / 32; ++kk) {
    const int kb = kk * 32;
    Frag16 a0, a1, b0, b1;
    // A 16x32 bf16, contiguous k order chosen here (B matches): lanes 0-15 hold K=0..15, lanes 16-31 hold K=16..31
    a0.h[0] = *(const v8bf*)(arow0 + kb + kh * 16);
    a0.h[1] = *(const v8bf*)(arow0 + kb + kh * 16 + 8);
    a1.h[0] = *(const v8bf*)(arow1 + kb + kh * 16);
    a1.h[1] = *(const v8bf*)(arow1 + kb + kh * 16 + 8);
    // B 32x16 bf16, same contiguous k order as A: lanes 0-15 hold K=0..15, lanes 16-31 hold K=16..31
    b0.h[0] = *(const v8bf*)(brow0 + kb + kh * 16);
    b0.h[1] = *(const v8bf*)(brow0 + kb + kh * 16 + 8);
    b1.h[0] = *(const v8bf*)(brow1 + kb + kh * 16);
    b1.h[1] = *(const v8bf*)(brow1 + kb + kh * 16 + 8);
    c00 = WMMA_BF16(a0, b0, c00);
    c01 = WMMA_BF16(a0, b1, c01);
    c10 = WMMA_BF16(a1, b0, c10);
    c11 = WMMA_BF16(a1, b1, c11);
  }

  store_tile(c00, mtile * 32 +  0, ntile * 32 +  0, l15, kh, scale, outb);
  store_tile(c01, mtile * 32 +  0, ntile * 32 + 16, l15, kh, scale, outb);
  store_tile(c10, mtile * 32 + 16, ntile * 32 +  0, l15, kh, scale, outb);
  store_tile(c11, mtile * 32 + 16, ntile * 32 + 16, l15, kh, scale, outb);
}

// ---------------------------------------------------------------------------
// Online-LSE helpers (per-lane registers only; scores arrive pre-scaled).
// ---------------------------------------------------------------------------
__device__ __forceinline__ void lse_update(const v8f& c, float& rm, float& rs) {
  float tm = fmaxf(fmaxf(fmaxf(c[0], c[1]), fmaxf(c[2], c[3])),
                   fmaxf(fmaxf(c[4], c[5]), fmaxf(c[6], c[7])));
  const float nm = fmaxf(rm, tm);
  float ss = 0.0f;
#pragma unroll
  for (int r = 0; r < 8; ++r) ss += __expf(c[r] - nm);
  rs = rs * __expf(rm - nm) + ss;
  rm = nm;
}

// Merge wave halves, remove the diagonal term, return logsumexp.
__device__ __forceinline__ float lse_final(float rm, float rs, float diag) {
  const float oM = __shfl_xor(rm, 16, 32);
  const float oS = __shfl_xor(rs, 16, 32);
  const float nm = fmaxf(rm, oM);
  float tot = rs * __expf(rm - nm) + oS * __expf(oM - nm);
  tot -= __expf(diag - nm);                  // self-exclusion (t == s)
  return nm + __logf(tot);
}

// ---------------------------------------------------------------------------
// Fused scores + masked logsumexp.  One wave per (b,h, 32-query tile).
// A-operand = key tile (16 keys x 64), B-operands = two fixed (beta-scaled)
// query tiles (D[m][n] = score(key m, query n)): each key fragment pair
// (4 b128 loads) feeds 4 WMMAs -> 1 load per WMMA.  The inner loop carries
// only max/exp/add; the diagonal term is subtracted once at the end.
// ---------------------------------------------------------------------------
__global__ void __launch_bounds__(128)
energy_attn(const bf16_t* __restrict__ qb,   // [B,H,S,D] bf16, beta-scaled
            const bf16_t* __restrict__ kb,   // [B,H,S,D] bf16
            const float* __restrict__ beta,  // [H]
            float* __restrict__ partials)    // [3072]
{
  const int lane = threadIdx.x & 31;
  const int wid  = threadIdx.x >> 5;
  const int gw   = blockIdx.x * 4 + wid;     // 0..3071
  const int qt   = gw & 63;                  // 32-query tile within (b,h)
  const int bh   = gw >> 6;                  // b*12 + h
  const int h    = bh % Hh;
  const float bt = beta[h];

  const bf16_t* qbh = qb + (size_t)bh * Ss * Dd;
  const bf16_t* kbh = kb + (size_t)bh * Ss * Dd;
  const int l15 = lane & 15;
  const int kh  = lane >> 4;

  // Fixed B operands: two query tiles of 16, two K-chunks of 32 each.
  const bf16_t* qrow0 = qbh + (size_t)(qt * 32 + l15) * Dd;
  const bf16_t* qrow1 = qrow0 + (size_t)16 * Dd;
  Frag16 bq0a, bq0b, bq1a, bq1b;
  bq0a.h[0] = *(const v8bf*)(qrow0 + kh * 16);
  bq0a.h[1] = *(const v8bf*)(qrow0 + kh * 16 + 8);
  bq0b.h[0] = *(const v8bf*)(qrow0 + 32 + kh * 16);
  bq0b.h[1] = *(const v8bf*)(qrow0 + 32 + kh * 16 + 8);
  bq1a.h[0] = *(const v8bf*)(qrow1 + kh * 16);
  bq1a.h[1] = *(const v8bf*)(qrow1 + kh * 16 + 8);
  bq1b.h[0] = *(const v8bf*)(qrow1 + 32 + kh * 16);
  bq1b.h[1] = *(const v8bf*)(qrow1 + 32 + kh * 16 + 8);

  const int q0 = qt * 32 + l15;              // this lane's query (tile 0)
  const int q1 = q0 + 16;                    // this lane's query (tile 1)

  // Diagonal scores s_qq = (beta*q_token) . k_token, via the resident Q
  // fragments and one extra K-row load per query tile (same lane offsets).
  float d0 = 0.0f, d1 = 0.0f;
  {
    const bf16_t* kr0 = kbh + (size_t)q0 * Dd;
    const bf16_t* kr1 = kbh + (size_t)q1 * Dd;
    Frag16 k0a, k0b, k1a, k1b;
    k0a.h[0] = *(const v8bf*)(kr0 + kh * 16);
    k0a.h[1] = *(const v8bf*)(kr0 + kh * 16 + 8);
    k0b.h[0] = *(const v8bf*)(kr0 + 32 + kh * 16);
    k0b.h[1] = *(const v8bf*)(kr0 + 32 + kh * 16 + 8);
    k1a.h[0] = *(const v8bf*)(kr1 + kh * 16);
    k1a.h[1] = *(const v8bf*)(kr1 + kh * 16 + 8);
    k1b.h[0] = *(const v8bf*)(kr1 + 32 + kh * 16);
    k1b.h[1] = *(const v8bf*)(kr1 + 32 + kh * 16 + 8);
#pragma unroll
    for (int i = 0; i < 16; ++i) {
      d0 += (float)bq0a.v[i] * (float)k0a.v[i] +
            (float)bq0b.v[i] * (float)k0b.v[i];
      d1 += (float)bq1a.v[i] * (float)k1a.v[i] +
            (float)bq1b.v[i] * (float)k1b.v[i];
    }
    d0 += __shfl_xor(d0, 16, 32);            // both wave halves -> full dot
    d1 += __shfl_xor(d1, 16, 32);
  }

  const float NEGBIG = -3.4e38f;
  float rm0 = NEGBIG, rs0 = 0.0f;
  float rm1 = NEGBIG, rs1 = 0.0f;

  for (int jt = 0; jt < Ss / 16; ++jt) {
    const bf16_t* krow = kbh + (size_t)(jt * 16 + l15) * Dd;
    Frag16 a0, a1;                           // keys, K-chunks 0..31 / 32..63
    a0.h[0] = *(const v8bf*)(krow + kh * 16);
    a0.h[1] = *(const v8bf*)(krow + kh * 16 + 8);
    a1.h[0] = *(const v8bf*)(krow + 32 + kh * 16);
    a1.h[1] = *(const v8bf*)(krow + 32 + kh * 16 + 8);

    v8f c0 = {};
    c0 = WMMA_BF16(a0, bq0a, c0);
    c0 = WMMA_BF16(a1, bq0b, c0);
    v8f c1 = {};
    c1 = WMMA_BF16(a0, bq1a, c1);
    c1 = WMMA_BF16(a1, bq1b, c1);

    lse_update(c0, rm0, rs0);                // unmasked online logsumexp
    lse_update(c1, rm1, rs1);
  }

  // Merge halves, remove diagonal, sum this wave's 32 query contributions.
  float contrib = (lse_final(rm0, rs0, d0) + lse_final(rm1, rs1, d1)) / bt;
#pragma unroll
  for (int off = 1; off < 16; off <<= 1)
    contrib += __shfl_xor(contrib, off, 32);

  if (lane == 0) partials[gw] = -contrib;    // energy contribution
}

// ---------------------------------------------------------------------------
// Deterministic fixed-order reduction of the wave partials.
// ---------------------------------------------------------------------------
__global__ void reduce_partials(const float* __restrict__ p,
                                float* __restrict__ out, int n) {
  __shared__ float sm[256];
  float s = 0.0f;
  for (int i = threadIdx.x; i < n; i += 256) s += p[i];
  sm[threadIdx.x] = s;
  __syncthreads();
  for (int off = 128; off > 0; off >>= 1) {
    if ((int)threadIdx.x < off) sm[threadIdx.x] += sm[threadIdx.x + off];
    __syncthreads();
  }
  if (threadIdx.x == 0) out[0] = sm[0];
}

// ---------------------------------------------------------------------------
extern "C" void kernel_launch(void* const* d_in, const int* in_sizes, int n_in,
                              void* d_out, int out_size, void* d_ws, size_t ws_size,
                              hipStream_t stream) {
  const float* x    = (const float*)d_in[0];
  const float* wq   = (const float*)d_in[1];
  const float* wk   = (const float*)d_in[2];
  const float* beta = (const float*)d_in[3];

  char* ws = (char*)d_ws;
  bf16_t* xb   = (bf16_t*)(ws + OFF_XB);
  bf16_t* wqb  = (bf16_t*)(ws + OFF_WQB);
  bf16_t* wkb  = (bf16_t*)(ws + OFF_WKB);
  bf16_t* qbuf = (bf16_t*)(ws + OFF_QB);
  bf16_t* kbuf = (bf16_t*)(ws + OFF_KB);
  float*  part = (float*)(ws + OFF_PART);

  // 1) f32 -> bf16 conversions
  {
    int n4 = Mrows * Ee / 4;
    cvt_f32_to_bf16_x4<<<(n4 + 255) / 256, 256, 0, stream>>>(x, xb, n4);
    int w4 = Ncols * Ee / 4;
    cvt_f32_to_bf16_x4<<<(w4 + 255) / 256, 256, 0, stream>>>(wq, wqb, w4);
    cvt_f32_to_bf16_x4<<<(w4 + 255) / 256, 256, 0, stream>>>(wk, wkb, w4);
  }

  // 2) Q (beta-scaled) and K projections: bf16 WMMA, f32 accum
  {
    dim3 grid(Mrows / 128, Ncols / 32);   // 64 x 24 blocks, 4 waves/block
    proj_gemm<<<grid, 128, 0, stream>>>(xb, wqb, beta, qbuf);
    proj_gemm<<<grid, 128, 0, stream>>>(xb, wkb, nullptr, kbuf);
  }

  // 3) Fused scores + LSE (one wave per 32-query tile)
  energy_attn<<<NWAVES_ATTN / 4, 128, 0, stream>>>(qbuf, kbuf, beta, part);

  // 4) Deterministic scalar reduction into d_out
  reduce_partials<<<1, 256, 0, stream>>>(part, (float*)d_out, NWAVES_ATTN);
}
